// AttentionHead_48911087567153
// MI455X (gfx1250) — matrix operand order repaired and checked by reading, not compiled, not run on hardware
//
#include <hip/hip_runtime.h>

#define DEV static __device__ __forceinline__

typedef __attribute__((ext_vector_type(16))) __bf16 v16bf;
typedef __attribute__((ext_vector_type(8)))  float  v8f;

// ---------------- problem constants ----------------
constexpr int B_ = 4, S_ = 2048, DM_ = 1024, DH_ = 64, K2_ = 1024;
constexpr int NQROW = B_ * S_;    // 8192 rows for q/k/v
constexpr int NRROW = B_ * K2_;   // 4096 rows for qr/kr

// ---------------- workspace layout (ushort elements) ----------------
constexpr size_t N_WT = (size_t)5 * DM_ * DH_;       // 327680 per (hi|lo)
constexpr size_t N_Q  = (size_t)NQROW * DH_;         // 524288
constexpr size_t N_R  = (size_t)NRROW * DH_;         // 262144
constexpr size_t OFF_WTH = 0;
constexpr size_t OFF_WTL = OFF_WTH + N_WT;
constexpr size_t OFF_QH  = OFF_WTL + N_WT;
constexpr size_t OFF_QL  = OFF_QH  + N_Q;
constexpr size_t OFF_KH  = OFF_QL  + N_Q;
constexpr size_t OFF_KL  = OFF_KH  + N_Q;
constexpr size_t OFF_VTH = OFF_KL  + N_Q;   // v transposed: (B*64, 2048)
constexpr size_t OFF_VTL = OFF_VTH + N_Q;
constexpr size_t OFF_QRH = OFF_VTL + N_Q;
constexpr size_t OFF_QRL = OFF_QRH + N_R;
constexpr size_t OFF_KRH = OFF_QRL + N_R;
constexpr size_t OFF_KRL = OFF_KRH + N_R;

// ---------------- bf16 helpers ----------------
DEV unsigned short f2bf(float f) {
    unsigned int u = __float_as_uint(f);
    unsigned int r = u + 0x7fffu + ((u >> 16) & 1u);   // RNE
    return (unsigned short)(r >> 16);
}
DEV float bf2f(unsigned short h) {
    return __uint_as_float(((unsigned int)h) << 16);
}
DEV void split_bf(float f, unsigned short& hi, unsigned short& lo) {
    hi = f2bf(f);
    lo = f2bf(f - bf2f(hi));
}
DEV unsigned int packsplit(float p) {
    unsigned short h = f2bf(p);
    unsigned short l = f2bf(p - bf2f(h));
    return (unsigned int)h | ((unsigned int)l << 16);
}
DEV int iclamp(int x, int lo, int hi) { return x < lo ? lo : (x > hi ? hi : x); }

union FragB16 { v16bf v; unsigned short u[16]; uint4 q[2]; };
union FragC   { v8f   v; float f[8]; };

// split (hi+lo) 3-product WMMA accumulate: C += A*B with bf16x3 emulation
DEV v8f wmma3(v16bf ah, v16bf al, v16bf bh, v16bf bl, v8f c) {
    c = __builtin_amdgcn_wmma_f32_16x16x32_bf16(false, ah, false, bh, (short)0, c, false, false);
    c = __builtin_amdgcn_wmma_f32_16x16x32_bf16(false, ah, false, bl, (short)0, c, false, false);
    c = __builtin_amdgcn_wmma_f32_16x16x32_bf16(false, al, false, bh, (short)0, c, false, false);
    return c;
}

// B-fragment: 16 contiguous bf16 (lane n = lane%16, K chunk selected by half)
DEV v16bf load_frag16(const unsigned short* p) {
    FragB16 f;
    f.q[0] = *(const uint4*)(p);
    f.q[1] = *(const uint4*)(p + 8);
    return f.v;
}
// A-fragment: two 8-element contiguous bf16 chunks
DEV v16bf load_fragA(const unsigned short* p0, const unsigned short* p1) {
    FragB16 f;
    f.q[0] = *(const uint4*)(p0);
    f.q[1] = *(const uint4*)(p1);
    return f.v;
}
// A-fragment built from fp32 memory, split into hi/lo frags
DEV void fragA_f32(const float* p0, const float* p1, v16bf& ah, v16bf& al) {
    FragB16 H, L;
#pragma unroll
    for (int i = 0; i < 8; i++) { unsigned short h, l; split_bf(p0[i], h, l); H.u[i] = h; L.u[i] = l; }
#pragma unroll
    for (int i = 0; i < 8; i++) { unsigned short h, l; split_bf(p1[i], h, l); H.u[8 + i] = h; L.u[8 + i] = l; }
    ah = H.v; al = L.v;
}

// =====================================================================
// Kernel 1: transpose + split weights -> Wt[hi|lo][w][n][k]  (n-major)
// =====================================================================
__global__ void prep_w_kernel(const float* __restrict__ Wq, const float* __restrict__ Wk,
                              const float* __restrict__ Wv, const float* __restrict__ Wqr,
                              const float* __restrict__ Wkr, unsigned short* __restrict__ ws) {
    unsigned int gid = blockIdx.x * blockDim.x + threadIdx.x;   // 0 .. 5*65536-1
    unsigned int w = gid >> 16;
    unsigned int r = gid & 0xffffu;           // dest index: n*1024 + k
    unsigned int n = r >> 10;
    unsigned int k = r & 1023u;
    const float* W = (w == 0) ? Wq : (w == 1) ? Wk : (w == 2) ? Wv : (w == 3) ? Wqr : Wkr;
    float val = W[k * 64 + n];
    unsigned short h, l;
    split_bf(val, h, l);
    ws[OFF_WTH + (size_t)w * 65536 + r] = h;
    ws[OFF_WTL + (size_t)w * 65536 + r] = l;
}

// =====================================================================
// Kernel 2: projections  Y = X @ W + b  (bf16x3 WMMA), write hi/lo
// wave tiles: p=0..2 over 8192 rows (q,k,v), p=3..4 over 4096 rows (qr,kr)
// =====================================================================
__global__ void proj_kernel(const float* __restrict__ x, const float* __restrict__ pos_x,
                            const float* __restrict__ bq, const float* __restrict__ bk,
                            const float* __restrict__ bv, const float* __restrict__ bqr,
                            const float* __restrict__ bkr, unsigned short* __restrict__ ws) {
    int wave = (blockIdx.x * blockDim.x + threadIdx.x) >> 5;   // 0..2047
    int lane = threadIdx.x & 31;
    int hf = lane >> 4;       // half of wave
    int ln = lane & 15;

    int p, tile;
    if (wave < 1536) { p = wave / 512; tile = wave % 512; }
    else             { p = 3 + (wave - 1536) / 256; tile = (wave - 1536) % 256; }

    const float* X = (p < 3) ? x : pos_x;
    const float* bias = (p == 0) ? bq : (p == 1) ? bk : (p == 2) ? bv : (p == 3) ? bqr : bkr;
    const unsigned short* wh = ws + OFF_WTH + (size_t)p * 65536;
    const unsigned short* wl = ws + OFF_WTL + (size_t)p * 65536;

    int row0 = tile * 16;
    const float* xrow = X + (size_t)(row0 + ln) * DM_;   // A row m = ln

    FragC C[4];
#pragma unroll
    for (int nt = 0; nt < 4; nt++)
#pragma unroll
        for (int i = 0; i < 8; i++) C[nt].f[i] = 0.0f;

    for (int kb = 0; kb < DM_; kb += 32) {
        v16bf ah, al;
        fragA_f32(xrow + kb + 16 * hf, xrow + kb + 16 * hf + 8, ah, al);
#pragma unroll
        for (int nt = 0; nt < 4; nt++) {
            size_t wo = (size_t)(nt * 16 + ln) * DM_ + kb + 16 * hf;
            v16bf bh = load_frag16(wh + wo);
            v16bf bl = load_frag16(wl + wo);
            C[nt].v = wmma3(ah, al, bh, bl, C[nt].v);
        }
    }

    unsigned short* q_hi  = ws + OFF_QH;  unsigned short* q_lo  = ws + OFF_QL;
    unsigned short* k_hi  = ws + OFF_KH;  unsigned short* k_lo  = ws + OFF_KL;
    unsigned short* vt_hi = ws + OFF_VTH; unsigned short* vt_lo = ws + OFF_VTL;
    unsigned short* qr_hi = ws + OFF_QRH; unsigned short* qr_lo = ws + OFF_QRL;
    unsigned short* kr_hi = ws + OFF_KRH; unsigned short* kr_lo = ws + OFF_KRL;

#pragma unroll
    for (int nt = 0; nt < 4; nt++) {
        int d = nt * 16 + ln;
        float bia = bias[d];
#pragma unroll
        for (int v = 0; v < 8; v++) {
            int row = row0 + v + 8 * hf;               // C-layout row
            float val = C[nt].f[v] + bia;
            unsigned short h, l;
            split_bf(val, h, l);
            if (p == 0)      { q_hi[(size_t)row * 64 + d] = h;  q_lo[(size_t)row * 64 + d] = l; }
            else if (p == 1) { k_hi[(size_t)row * 64 + d] = h;  k_lo[(size_t)row * 64 + d] = l; }
            else if (p == 2) {
                int b = row >> 11, s = row & 2047;     // transpose v: (b*64+d, s)
                size_t o = (size_t)(b * 64 + d) * 2048 + s;
                vt_hi[o] = h; vt_lo[o] = l;
            }
            else if (p == 3) { qr_hi[(size_t)row * 64 + d] = h; qr_lo[(size_t)row * 64 + d] = l; }
            else             { kr_hi[(size_t)row * 64 + d] = h; kr_lo[(size_t)row * 64 + d] = l; }
        }
    }
}

// =====================================================================
// Kernel 3: fused disentangled attention (flash online softmax, split-K)
// one BLOCK per (batch, 16-query tile); 4 waves partition the 64
// 32-key blocks (strided), partial (m,l,O) merged via LDS + barrier.
// =====================================================================
__global__ void attn_kernel(const unsigned short* __restrict__ ws,
                            const int* __restrict__ mask_all,
                            float* __restrict__ out) {
    __shared__ float        lds_c2p[4][16 * 48];
    __shared__ float        lds_p2c[4][48 * 32];
    __shared__ unsigned int lds_p  [4][16 * 32];
    __shared__ float        cb_O[4][16][64];
    __shared__ float        cb_m[4][16];
    __shared__ float        cb_l[4][16];

    int waveId = threadIdx.x >> 5;
    int lane = threadIdx.x & 31;
    int hf = lane >> 4;
    int ln = lane & 15;

    int tile = blockIdx.x;                   // 0..511
    int b = tile >> 7;                       // tile / 128
    int i0 = (tile & 127) << 4;

    const unsigned short* q_hi  = ws + OFF_QH;  const unsigned short* q_lo  = ws + OFF_QL;
    const unsigned short* k_hi  = ws + OFF_KH;  const unsigned short* k_lo  = ws + OFF_KL;
    const unsigned short* vt_hi = ws + OFF_VTH; const unsigned short* vt_lo = ws + OFF_VTL;
    const unsigned short* qr_hi = ws + OFF_QRH; const unsigned short* qr_lo = ws + OFF_QRL;
    const unsigned short* kr_hi = ws + OFF_KRH; const unsigned short* kr_lo = ws + OFF_KRL;
    const int* M = mask_all + (size_t)b * S_ * S_;

    float* c2pS = lds_c2p[waveId];
    float* p2cS = lds_p2c[waveId];
    unsigned int* Pb = lds_p[waveId];

    // ---- Q A-fragments (kept resident): m = ln, rows i0..i0+15 ----
    v16bf qa_h[2], qa_l[2];
    {
        size_t qb = (size_t)(b * S_ + i0 + ln) * 64;
#pragma unroll
        for (int kc = 0; kc < 2; kc++) {
            int kb = kc * 32;
            qa_h[kc] = load_fragA(q_hi + qb + kb + 16 * hf, q_hi + qb + kb + 16 * hf + 8);
            qa_l[kc] = load_fragA(q_lo + qb + kb + 16 * hf, q_lo + qb + kb + 16 * hf + 8);
        }
    }

    FragC O[4];
#pragma unroll
    for (int nt = 0; nt < 4; nt++)
#pragma unroll
        for (int i = 0; i < 8; i++) O[nt].f[i] = 0.0f;
    float mrow[8], lrow[8];
#pragma unroll
    for (int v = 0; v < 8; v++) { mrow[v] = -1e30f; lrow[v] = 0.0f; }

    const float rscale = 0.07216878364870323f;   // 1/sqrt(3*DH)

    for (int jb = waveId; jb < 64; jb += 4) {
        int j0 = jb * 32;
        int r0 = i0 - j0 + 512;

        // prefetch next iteration's k / v lines (global_prefetch_b8)
        if (jb + 4 < 64) {
            int j0n = j0 + 128;
            __builtin_prefetch(k_hi + (size_t)(b * S_ + j0n + ln) * 64, 0, 3);
            __builtin_prefetch(vt_hi + (size_t)(b * 64 + ln) * 2048 + j0n, 0, 3);
        }

        // ---- c2p slab: tile16x48 = q @ kr_slab^T ; slab col t, row(t)=clamp(r0-31+t) ----
#pragma unroll
        for (int tb = 0; tb < 48; tb += 16) {
            int t = tb + ln;                               // B-frag col n = ln
            int rr = iclamp(r0 - 31 + t, 0, K2_ - 1);
            size_t kro = (size_t)(b * K2_ + rr) * 64;
            FragC Cc;
#pragma unroll
            for (int i = 0; i < 8; i++) Cc.f[i] = 0.0f;
#pragma unroll
            for (int kc = 0; kc < 2; kc++) {
                int kb = kc * 32;
                v16bf bh = load_frag16(kr_hi + kro + kb + 16 * hf);
                v16bf bl = load_frag16(kr_lo + kro + kb + 16 * hf);
                Cc.v = wmma3(qa_h[kc], qa_l[kc], bh, bl, Cc.v);
            }
#pragma unroll
            for (int v = 0; v < 8; v++)
                c2pS[(v + 8 * hf) * 48 + t] = Cc.f[v];
        }

        // ---- per 16-key sub-block: c2c + p2c ----
        FragC Sc[2];
#pragma unroll
        for (int jj = 0; jj < 2; jj++) {
            int j0s = j0 + jj * 16;
            size_t ko = (size_t)(b * S_ + j0s + ln) * 64;   // B-frag n = ln
            v16bf kbh[2], kbl[2];
#pragma unroll
            for (int kc = 0; kc < 2; kc++) {
                kbh[kc] = load_frag16(k_hi + ko + kc * 32 + 16 * hf);
                kbl[kc] = load_frag16(k_lo + ko + kc * 32 + 16 * hf);
            }
            // c2c
#pragma unroll
            for (int i = 0; i < 8; i++) Sc[jj].f[i] = 0.0f;
#pragma unroll
            for (int kc = 0; kc < 2; kc++)
                Sc[jj].v = wmma3(qa_h[kc], qa_l[kc], kbh[kc], kbl[kc], Sc[jj].v);
            // p2c: qr_slab(48x64) @ k_sub^T(64x16) -> 48x16, rows t
#pragma unroll
            for (int ta = 0; ta < 48; ta += 16) {
                int t = ta + ln;                           // A-frag row m = ln
                int rr = iclamp(r0 - 31 + t, 0, K2_ - 1);
                size_t qo = (size_t)(b * K2_ + rr) * 64;
                FragC Cp;
#pragma unroll
                for (int i = 0; i < 8; i++) Cp.f[i] = 0.0f;
#pragma unroll
                for (int kc = 0; kc < 2; kc++) {
                    int kb = kc * 32;
                    v16bf ah = load_fragA(qr_hi + qo + kb + 16 * hf, qr_hi + qo + kb + 16 * hf + 8);
                    v16bf al = load_fragA(qr_lo + qo + kb + 16 * hf, qr_lo + qo + kb + 16 * hf + 8);
                    Cp.v = wmma3(ah, al, kbh[kc], kbl[kc], Cp.v);
                }
#pragma unroll
                for (int v = 0; v < 8; v++)
                    p2cS[(ta + v + 8 * hf) * 32 + jj * 16 + ln] = Cp.f[v];
            }
        }
        asm volatile("s_wait_dscnt 0" ::: "memory");

        // ---- combine scores + mask (anti-diagonal gather from LDS) ----
#pragma unroll
        for (int jj = 0; jj < 2; jj++) {
#pragma unroll
            for (int v = 0; v < 8; v++) {
                int di = v + 8 * hf;
                int dj = jj * 16 + ln;
                int t = 31 + di - dj;                      // clamp folded into slab rows
                float s = (Sc[jj].f[v] + c2pS[di * 48 + t] + p2cS[t * 32 + dj]) * rscale;
                int mv = M[(size_t)(i0 + di) * S_ + j0 + dj];
                if (mv == 0) s = -1e30f;
                Sc[jj].f[v] = s;
            }
        }

        // ---- online softmax update over this 32-key block ----
        float corr[8];
#pragma unroll
        for (int v = 0; v < 8; v++) {
            float a = fmaxf(Sc[0].f[v], Sc[1].f[v]);
#pragma unroll
            for (int msk = 1; msk < 16; msk <<= 1)
                a = fmaxf(a, __shfl_xor(a, msk, 32));
            float mn = fmaxf(mrow[v], a);
            float c = __expf(mrow[v] - mn);
            float p0 = __expf(Sc[0].f[v] - mn);
            float p1 = __expf(Sc[1].f[v] - mn);
            float rs = p0 + p1;
#pragma unroll
            for (int msk = 1; msk < 16; msk <<= 1)
                rs += __shfl_xor(rs, msk, 32);
            lrow[v] = lrow[v] * c + rs;
            mrow[v] = mn;
            corr[v] = c;
            int di = v + 8 * hf;
            Pb[di * 32 + ln]      = packsplit(p0);
            Pb[di * 32 + 16 + ln] = packsplit(p1);
        }
#pragma unroll
        for (int nt = 0; nt < 4; nt++)
#pragma unroll
            for (int v = 0; v < 8; v++) O[nt].f[v] *= corr[v];
        asm volatile("s_wait_dscnt 0" ::: "memory");

        // ---- PV: O += P(16x32) @ V(32x64) ----
        v16bf pah, pal;
        {
            FragB16 H, L;
#pragma unroll
            for (int r = 0; r < 8; r++) {
                int K = 16 * hf + 2 * r;
                unsigned int u0 = Pb[ln * 32 + K];
                unsigned int u1 = Pb[ln * 32 + K + 1];
                H.u[2 * r]     = (unsigned short)(u0 & 0xffffu);
                H.u[2 * r + 1] = (unsigned short)(u1 & 0xffffu);
                L.u[2 * r]     = (unsigned short)(u0 >> 16);
                L.u[2 * r + 1] = (unsigned short)(u1 >> 16);
            }
            pah = H.v; pal = L.v;
        }
#pragma unroll
        for (int nt = 0; nt < 4; nt++) {
            size_t vo = (size_t)(b * 64 + nt * 16 + ln) * 2048 + j0 + 16 * hf;
            v16bf vh = load_frag16(vt_hi + vo);
            v16bf vl = load_frag16(vt_lo + vo);
            O[nt].v = wmma3(pah, pal, vh, vl, O[nt].v);
        }
    }

    // ---- stash per-wave partial state ----
#pragma unroll
    for (int nt = 0; nt < 4; nt++)
#pragma unroll
        for (int v = 0; v < 8; v++)
            cb_O[waveId][v + 8 * hf][nt * 16 + ln] = O[nt].f[v];
    if (ln == 0) {
#pragma unroll
        for (int v = 0; v < 8; v++) {
            cb_m[waveId][v + 8 * hf] = mrow[v];
            cb_l[waveId][v + 8 * hf] = lrow[v];
        }
    }
    __syncthreads();

    // ---- merge 4 split-K partial softmax states, store fp32 ----
    int tid = threadIdx.x;
#pragma unroll
    for (int i = 0; i < 8; i++) {
        int idx = tid + 128 * i;            // 0..1023 over 16x64 tile
        int r = idx >> 6, d = idx & 63;
        float m0 = cb_m[0][r], m1 = cb_m[1][r], m2 = cb_m[2][r], m3 = cb_m[3][r];
        float ms = fmaxf(fmaxf(m0, m1), fmaxf(m2, m3));
        float e0 = __expf(m0 - ms), e1 = __expf(m1 - ms);
        float e2 = __expf(m2 - ms), e3 = __expf(m3 - ms);
        float lt = cb_l[0][r] * e0 + cb_l[1][r] * e1 + cb_l[2][r] * e2 + cb_l[3][r] * e3;
        float ot = cb_O[0][r][d] * e0 + cb_O[1][r][d] * e1
                 + cb_O[2][r][d] * e2 + cb_O[3][r][d] * e3;
        float inv = (lt > 0.0f) ? (1.0f / lt) : 0.0f;
        out[(size_t)(b * S_ + i0 + r) * 64 + d] = ot * inv;
    }
}

// =====================================================================
extern "C" void kernel_launch(void* const* d_in, const int* in_sizes, int n_in,
                              void* d_out, int out_size, void* d_ws, size_t ws_size,
                              hipStream_t stream) {
    const float* x    = (const float*)d_in[0];
    const float* posx = (const float*)d_in[1];
    const int*   mask = (const int*)d_in[2];
    const float* Wq  = (const float*)d_in[3];  const float* bq  = (const float*)d_in[4];
    const float* Wk  = (const float*)d_in[5];  const float* bk  = (const float*)d_in[6];
    const float* Wv  = (const float*)d_in[7];  const float* bv  = (const float*)d_in[8];
    const float* Wqr = (const float*)d_in[9];  const float* bqr = (const float*)d_in[10];
    const float* Wkr = (const float*)d_in[11]; const float* bkr = (const float*)d_in[12];
    unsigned short* ws = (unsigned short*)d_ws;
    float* out = (float*)d_out;

    // 1) weight transpose + bf16 hi/lo split: 5*64*1024 = 327680 elements
    prep_w_kernel<<<dim3(1280), dim3(256), 0, stream>>>(Wq, Wk, Wv, Wqr, Wkr, ws);
    // 2) projections: 2048 wave-tiles, 4 waves/block
    proj_kernel<<<dim3(512), dim3(128), 0, stream>>>(x, posx, bq, bk, bv, bqr, bkr, ws);
    // 3) fused attention: 512 q-tile blocks x 4 split-K waves = 2048 waves
    attn_kernel<<<dim3(512), dim3(128), 0, stream>>>(ws, mask, out);
}
